// LSTM_Model_19310172962819
// MI455X (gfx1250) — matrix operand order repaired and checked by reading, not compiled, not run on hardware
//
#include <hip/hip_runtime.h>
#include <hip/hip_bf16.h>

// ---------------------------------------------------------------------------
// 2-layer LSTM (B=128, T=256, H=512, E=300) for MI455X / gfx1250.
// GEMMs run on v_wmma_f32_16x16x32_bf16 (bf16 operands, f32 accumulate).
// Weights (7.5MB as bf16) live in L2 (192MB) -> recurrent loop is L2-resident.
// ---------------------------------------------------------------------------

typedef __attribute__((ext_vector_type(16))) __bf16          v16bf;
typedef __attribute__((ext_vector_type(8)))  float           v8f;
typedef __attribute__((ext_vector_type(8)))  unsigned short  ushort8;

union FragU {              // 32 bytes: one WMMA bf16 operand fragment per lane
    v16bf   bf;
    ushort8 h[2];
};

__device__ __forceinline__ unsigned short f2bf(float f) {
    unsigned int u = __builtin_bit_cast(unsigned int, f);
    unsigned int r = u + 0x7fffu + ((u >> 16) & 1u);   // round-to-nearest-even
    if ((u & 0x7f800000u) == 0x7f800000u) r = u;       // keep NaN/Inf
    return (unsigned short)(r >> 16);
}

__device__ __forceinline__ float sigmoidf_(float x) {
    return 1.0f / (1.0f + __expf(-x));
}

// ---------------------------------------------------------------------------
// Zero a float buffer (states must be re-zeroed every call: graph replays).
// ---------------------------------------------------------------------------
__global__ void zero_f32(float* __restrict__ p, int n) {
    for (int i = blockIdx.x * blockDim.x + threadIdx.x; i < n;
         i += gridDim.x * blockDim.x)
        p[i] = 0.0f;
}

// ---------------------------------------------------------------------------
// Embedding gather + f32->bf16, K padded 300 -> 320 (multiple of 32).
// Xall layout: [t][b][320] bf16.  x is [B][T].
// ---------------------------------------------------------------------------
__global__ void embed_gather(const int* __restrict__ x,
                             const float* __restrict__ emb,
                             unsigned short* __restrict__ Xall,
                             int T, int B, int E, int Kx) {
    long total = (long)T * B * Kx;
    for (long e = (long)blockIdx.x * blockDim.x + threadIdx.x; e < total;
         e += (long)gridDim.x * blockDim.x) {
        int  k   = (int)(e % Kx);
        long row = e / Kx;
        int  b   = (int)(row % B);
        int  t   = (int)(row / B);
        float val = 0.0f;
        if (k < E) {
            int idx = x[(long)b * T + t];
            val = emb[(long)idx * E + k];
        }
        Xall[e] = f2bf(val);
    }
}

// ---------------------------------------------------------------------------
// Swizzle f32 weights into WMMA B-fragment layout (ISA 7.12.2, bf16 32x16):
//   fragment(kb, nt) = 512 bf16, addressed as lane*16 + v where
//   K_local = (lane>=16 ? 8 : 0) + (v & 7) + (v>=8 ? 16 : 0),  N_local = lane & 15.
// Rows beyond `valid` are zero-padded (covers E=300 -> 320 padding).
// ---------------------------------------------------------------------------
__global__ void swizzle_w(const float* __restrict__ W, int row_off, int valid,
                          int KB, int ncols, unsigned short* __restrict__ dst) {
    long total = (long)KB * 128 * 512;
    for (long e = (long)blockIdx.x * blockDim.x + threadIdx.x; e < total;
         e += (long)gridDim.x * blockDim.x) {
        int pos  = (int)(e & 511);
        int nt   = (int)((e >> 9) & 127);
        int kb   = (int)(e >> 16);
        int lane = pos >> 4;
        int v    = pos & 15;
        int kloc = ((lane & 16) >> 1) + (v & 7) + ((v & 8) << 1);  // lane>=16 -> +8, v>=8 -> +16
        int rloc = kb * 32 + kloc;
        int col  = nt * 16 + (lane & 15);
        float val = (rloc < valid) ? W[(long)(row_off + rloc) * ncols + col]
                                   : 0.0f;
        dst[e] = f2bf(val);
    }
}

// ---------------------------------------------------------------------------
// One K-block: load A fragment (row-major bf16, ISA A-layout: lane = row,
// K-halves selected by lane[4]) + 4 pre-swizzled B fragments, issue 4 WMMAs.
// ---------------------------------------------------------------------------
__device__ __forceinline__ void wmma_kblock(const unsigned short* ap, int kSel,
                                            const unsigned short* wp,
                                            bool prefetchNext,
                                            v8f& acc0, v8f& acc1,
                                            v8f& acc2, v8f& acc3) {
    FragU a;
    a.h[0] = *(const ushort8*)(ap + kSel);        // K = base + 0..7
    a.h[1] = *(const ushort8*)(ap + kSel + 16);   // K = base + 16..23

    if (prefetchNext)                              // next K-block's weights:
        __builtin_prefetch(wp + 128 * 512, 0, 3);  // WGP-scope prefetch

    FragU b0, b1, b2, b3;
    b0.h[0] = *(const ushort8*)(wp);          b0.h[1] = *(const ushort8*)(wp + 8);
    b1.h[0] = *(const ushort8*)(wp + 512);    b1.h[1] = *(const ushort8*)(wp + 520);
    b2.h[0] = *(const ushort8*)(wp + 1024);   b2.h[1] = *(const ushort8*)(wp + 1032);
    b3.h[0] = *(const ushort8*)(wp + 1536);   b3.h[1] = *(const ushort8*)(wp + 1544);

    acc0 = __builtin_amdgcn_wmma_f32_16x16x32_bf16(false, a.bf, false, b0.bf,
                                                   (short)0, acc0, false, false);
    acc1 = __builtin_amdgcn_wmma_f32_16x16x32_bf16(false, a.bf, false, b1.bf,
                                                   (short)0, acc1, false, false);
    acc2 = __builtin_amdgcn_wmma_f32_16x16x32_bf16(false, a.bf, false, b2.bf,
                                                   (short)0, acc2, false, false);
    acc3 = __builtin_amdgcn_wmma_f32_16x16x32_bf16(false, a.bf, false, b3.bf,
                                                   (short)0, acc3, false, false);
}

// ---------------------------------------------------------------------------
// Gate GEMM: gates[128][2048] = [A0 | A1] * W, K = 32*KB, A0 covers the first
// KB0 K-blocks.  K-loop is split per A-segment so the steady-state inner loop
// has no pointer select (no cndmask / WMMA->VALU hazard NOPs).
// One wave = one 16-row tile x 4 column tiles (4 independent WMMA chains).
// Grid: 64 blocks x 128 threads (4 waves).
// ---------------------------------------------------------------------------
__global__ void __launch_bounds__(128)
gemm_gates(const unsigned short* __restrict__ A0, int lda0, int KB0,
           const unsigned short* __restrict__ A1, int lda1,
           const unsigned short* __restrict__ Wfrag, int KB,
           float* __restrict__ gates) {
    const int lane   = threadIdx.x & 31;
    const int wave   = threadIdx.x >> 5;
    const int mt     = blockIdx.x >> 3;                    // 0..7
    const int ntBase = ((blockIdx.x & 7) * 4 + wave) * 4;  // 0..124 step 4
    const int kSel   = (lane >> 4) * 8;    // A-fragment K-half per ISA layout

    const unsigned short* arow0 = A0 + (long)(mt * 16 + (lane & 15)) * lda0;
    const unsigned short* arow1 = A1 + (long)(mt * 16 + (lane & 15)) * lda1;
    const unsigned short* wp =
        Wfrag + (long)ntBase * 512 + lane * 16;

    v8f acc0 = {}, acc1 = {}, acc2 = {}, acc3 = {};

#pragma unroll 2
    for (int kb = 0; kb < KB0; ++kb) {
        wmma_kblock(arow0 + kb * 32, kSel, wp, kb + 1 < KB,
                    acc0, acc1, acc2, acc3);
        wp += 128 * 512;
    }
#pragma unroll 2
    for (int kb = KB0; kb < KB; ++kb) {
        wmma_kblock(arow1 + (kb - KB0) * 32, kSel, wp, kb + 1 < KB,
                    acc0, acc1, acc2, acc3);
        wp += 128 * 512;
    }

    // C layout (f32 16x16): VGPR r -> row r (lanes 0-15) / row r+8 (lanes 16-31)
    const int rBase = mt * 16 + ((lane >> 4) << 3);
    const int col   = ntBase * 16 + (lane & 15);
    float* g = gates + (long)rBase * 2048 + col;
#pragma unroll
    for (int r = 0; r < 8; ++r) {
        g[(long)r * 2048 +  0] = acc0[r];
        g[(long)r * 2048 + 16] = acc1[r];
        g[(long)r * 2048 + 32] = acc2[r];
        g[(long)r * 2048 + 48] = acc3[r];
    }
}

// ---------------------------------------------------------------------------
// LSTM cell elementwise (gate order i,j,f,o; forget_bias = 1.0).
// Writes f32 cell state, bf16 h into hcat[:, hoff:hoff+512], optional f32 h.
// Grid: 256 blocks x 256 threads = 65536 = 128*512.
// ---------------------------------------------------------------------------
__global__ void lstm_cell(const float* __restrict__ gates,
                          const float* __restrict__ bias,
                          float* __restrict__ c,
                          unsigned short* __restrict__ hcat, int hoff,
                          float* __restrict__ hf32) {
    int idx = blockIdx.x * blockDim.x + threadIdx.x;
    int b = idx >> 9;
    int u = idx & 511;
    const float* g = gates + (long)b * 2048;
    float gi = g[u]        + bias[u];
    float gj = g[512 + u]  + bias[512 + u];
    float gf = g[1024 + u] + bias[1024 + u];
    float go = g[1536 + u] + bias[1536 + u];
    float cn = c[idx] * sigmoidf_(gf + 1.0f) + sigmoidf_(gi) * tanhf(gj);
    float hn = tanhf(cn) * sigmoidf_(go);
    c[idx] = cn;
    hcat[(long)b * 1024 + hoff + u] = f2bf(hn);
    if (hf32) hf32[idx] = hn;
}

// ---------------------------------------------------------------------------
// logits[128][2] = h1[128][512] * Wd[512][2] + bd   (tiny, plain f32)
// ---------------------------------------------------------------------------
__global__ void final_dense(const float* __restrict__ h,
                            const float* __restrict__ Wd,
                            const float* __restrict__ bd,
                            float* __restrict__ out) {
    int b = blockIdx.x * blockDim.x + threadIdx.x;
    if (b >= 128) return;
    float a0 = bd[0], a1 = bd[1];
    for (int u = 0; u < 512; ++u) {
        float hv = h[b * 512 + u];
        a0 += hv * Wd[u * 2 + 0];
        a1 += hv * Wd[u * 2 + 1];
    }
    out[b * 2 + 0] = a0;
    out[b * 2 + 1] = a1;
}

// ---------------------------------------------------------------------------
extern "C" void kernel_launch(void* const* d_in, const int* in_sizes, int n_in,
                              void* d_out, int out_size, void* d_ws, size_t ws_size,
                              hipStream_t stream) {
    const int*   x   = (const int*)  d_in[0];   // [B=128][T=256]
    const float* emb = (const float*)d_in[1];   // [50000][300]
    const float* W0  = (const float*)d_in[2];   // [812][2048]
    const float* b0  = (const float*)d_in[3];   // [2048]
    const float* W1  = (const float*)d_in[4];   // [1024][2048]
    const float* b1  = (const float*)d_in[5];   // [2048]
    const float* Wd  = (const float*)d_in[6];   // [512][2]
    const float* bd  = (const float*)d_in[7];   // [2]
    float* out = (float*)d_out;                 // [128][2]

    // ---- workspace carve (~32 MB), 256B aligned ----
    char* p = (char*)d_ws;
    auto carve = [&](size_t bytes) -> char* {
        char* r = p;
        p += (bytes + 255) & ~(size_t)255;
        return r;
    };
    unsigned short* Xall   = (unsigned short*)carve((size_t)256 * 128 * 320 * 2);
    unsigned short* W0frag = (unsigned short*)carve((size_t)26 * 128 * 512 * 2);
    unsigned short* W1frag = (unsigned short*)carve((size_t)32 * 128 * 512 * 2);
    float* gates0 = (float*)carve((size_t)128 * 2048 * 4);
    float* gates1 = (float*)carve((size_t)128 * 2048 * 4);
    float* c0     = (float*)carve((size_t)128 * 512 * 4);
    float* c1     = (float*)carve((size_t)128 * 512 * 4);
    unsigned short* hcat = (unsigned short*)carve((size_t)128 * 1024 * 2);
    float* h1f    = (float*)carve((size_t)128 * 512 * 4);

    // ---- reset recurrent state every call (graph replays reuse d_ws) ----
    zero_f32<<<256, 256, 0, stream>>>(c0, 128 * 512);
    zero_f32<<<256, 256, 0, stream>>>(c1, 128 * 512);
    zero_f32<<<256, 256, 0, stream>>>((float*)hcat, 128 * 1024 / 2);

    // ---- one-time prep: embeddings + weight fragment swizzle ----
    embed_gather<<<4096, 256, 0, stream>>>(x, emb, Xall, 256, 128, 300, 320);
    // W0 rows 0..299 (x part, padded to 320) -> kblocks 0..9
    swizzle_w<<<2560, 256, 0, stream>>>(W0, 0, 300, 10, 2048, W0frag);
    // W0 rows 300..811 (h part, K=512) -> kblocks 10..25
    swizzle_w<<<4096, 256, 0, stream>>>(W0, 300, 512, 16, 2048,
                                        W0frag + (size_t)10 * 128 * 512);
    // W1 rows 0..1023 -> kblocks 0..31
    swizzle_w<<<8192, 256, 0, stream>>>(W1, 0, 1024, 32, 2048, W1frag);

    // ---- recurrent loop: 256 steps x (gemm0, cell0, gemm1, cell1) ----
    for (int t = 0; t < 256; ++t) {
        // layer 0: A = [x_t (K=320, padded) | h0_prev (K=512)]
        gemm_gates<<<64, 128, 0, stream>>>(Xall + (size_t)t * 128 * 320, 320, 10,
                                           hcat, 1024, W0frag, 26, gates0);
        lstm_cell<<<256, 256, 0, stream>>>(gates0, b0, c0, hcat, 0, nullptr);
        // layer 1: A = [h0_new | h1_prev] = hcat (K=1024), second loop empty
        gemm_gates<<<64, 128, 0, stream>>>(hcat, 1024, 32,
                                           hcat, 1024, W1frag, 32, gates1);
        lstm_cell<<<256, 256, 0, stream>>>(gates1, b1, c1, hcat, 512, h1f);
    }

    final_dense<<<1, 128, 0, stream>>>(h1f, Wd, bd, out);
}
